// RNNModel_16037407883320
// MI455X (gfx1250) — matrix operand order repaired and checked by reading, not compiled, not run on hardware
//
#include <hip/hip_runtime.h>

// ---------------------------------------------------------------------------
// Problem constants (from the reference)
// ---------------------------------------------------------------------------
#define TT 128
#define BB 32
#define VV 50257
#define EE 512
#define HH 512
#define LL 2

typedef __attribute__((ext_vector_type(16))) __bf16 v16bf;
typedef __attribute__((ext_vector_type(8)))  __bf16 v8bf;
typedef __attribute__((ext_vector_type(8)))  float  v8f;
typedef __attribute__((ext_vector_type(4)))  unsigned int v4u;
typedef __attribute__((ext_vector_type(8)))  int v8i;
typedef __attribute__((ext_vector_type(4)))  int v4i;

static __device__ __forceinline__ __bf16 f2bf(float f) {
  union { float f; unsigned u; } x; x.f = f;
  unsigned r = x.u + 0x7FFFu + ((x.u >> 16) & 1u);   // round-to-nearest-even
  unsigned short s = (unsigned short)(r >> 16);
  union { unsigned short s; __bf16 b; } y; y.s = s;
  return y.b;
}

static __device__ __forceinline__ float sigmoidf_(float x) {
  return 1.0f / (1.0f + __expf(-x));
}

// ---------------------------------------------------------------------------
// Elementwise f32 -> bf16 cast
// ---------------------------------------------------------------------------
__global__ void cast_f32_bf16(const float* __restrict__ in,
                              __bf16* __restrict__ out, size_t n) {
  size_t i = (size_t)blockIdx.x * blockDim.x + threadIdx.x;
  if (i < n) out[i] = f2bf(in[i]);
}

// ---------------------------------------------------------------------------
// Embedding gather -> bf16 activations  x[(t*B+b), e] = emb[idx[t*B+b], e]
// ---------------------------------------------------------------------------
__global__ void embed_bf16(const int* __restrict__ idx,
                           const float* __restrict__ emb,
                           __bf16* __restrict__ x) {
  size_t i = (size_t)blockIdx.x * blockDim.x + threadIdx.x;
  size_t total = (size_t)TT * BB * EE;
  if (i >= total) return;
  int row = (int)(i / EE);
  int e   = (int)(i % EE);
  int id  = idx[row];
  if (id < 0) id = 0;
  if (id >= VV) id = VV - 1;
  x[i] = f2bf(emb[(size_t)id * EE + e]);
}

// ---------------------------------------------------------------------------
// Generic GEMM:  Out[M,N] = A[M,K] @ W[N,K]^T  (+ bias0[n] + bias1[n])
// bf16 inputs, f32 accumulate via v_wmma_f32_16x16x32_bf16.
//
// Block (8 waves) shares one 64-column W slab, staged into LDS via the
// Tensor Data Mover (one 64KB descriptor per block; OOB rows zero-filled by
// the D# tensor_dim bound).  Each wave computes a 32(M) x 64(N) patch:
// 2 M-tiles x 4 N-tiles, A fragments software-pipelined from global, B
// fragments from LDS.  Requires: M % 32 == 0, (M/32) % 8 == 0, K == 512.
// NT=true -> non-temporal stores (stream logits to HBM, keep dec_W in L2).
// ---------------------------------------------------------------------------
template <bool NT>
__global__ __launch_bounds__(256) void gemm_bf16_wmma(
    const __bf16* __restrict__ A,     // [M, K] row-major
    const __bf16* __restrict__ W,     // [N, K] row-major (B-matrix = W^T)
    const float*  __restrict__ bias0, // [N] or nullptr
    const float*  __restrict__ bias1, // [N] or nullptr
    float* __restrict__ Out,          // [M, N] row-major
    int M, int N, int K)
{
  __shared__ __align__(128) __bf16 wtile[64 * 512];   // 64KB W slab

  const int MP  = M / 32;
  int wv   = threadIdx.x >> 5;
  int lane = threadIdx.x & 31;
  int half = lane >> 4;                 // lane group (ISA A/B layouts)
  int lm   = lane & 15;

  int nPatch = (blockIdx.x * 8) / MP;   // uniform across the block (MP%8==0)
  int mPatch = (blockIdx.x * 8) % MP + wv;
  int n0   = nPatch * 64;
  int mrow = mPatch * 32;

  // ---- TDM: stage W[n0 .. n0+64) x K into LDS (zero-fill past N*K) ----
  if (threadIdx.x < 32) {               // wave 0 issues the descriptor
    unsigned long long ga =
        (unsigned long long)(const void*)(W + (size_t)n0 * K);
    unsigned lds_off    = (unsigned)(unsigned long long)(uintptr_t)&wtile[0];
    unsigned copy_elems = 64u * (unsigned)K;          // 32768 (fits 16 bits)
    long long remll     = (long long)(N - n0) * K;    // >= K > 0 always
    unsigned tdim0 = (remll < (long long)copy_elems) ? (unsigned)remll
                                                     : copy_elems;
    v4u g0;
    g0.x = 1u;                                        // count=1, user mode
    g0.y = lds_off;                                   // lds_addr [63:32]
    g0.z = (unsigned)(ga & 0xFFFFFFFFull);            // global_addr [95:64]
    g0.w = (unsigned)((ga >> 32) & 0x1FFFFFFull)      // global_addr [120:96]
           | (2u << 30);                              // type=2 ("image")
    v8i g1;
    g1[0] = (int)(1u << 16);                          // data_size = 2 bytes
    g1[1] = (int)((tdim0 & 0xFFFFu) << 16);           // tensor_dim0[15:0]
    g1[2] = (int)((tdim0 >> 16) | (1u << 16));        // dim0[31:16], dim1=1
    g1[3] = (int)((copy_elems & 0xFFFFu) << 16);      // tile_dim0 = 64*K
    g1[4] = 1;                                        // tile_dim1 = 1
    g1[5] = (int)copy_elems;                          // tensor_dim0_stride
    g1[6] = 0;
    g1[7] = 0;
    v4i gz4 = (v4i)0;                                 // groups 2/3 unused
    v8i gz8 = (v8i)0;
    __builtin_amdgcn_tensor_load_to_lds(g0, g1, gz4, gz4, gz8, 0);
    __builtin_amdgcn_s_wait_tensorcnt(0);
  }
  __syncthreads();

  v8f acc[2][4];
  for (int mt = 0; mt < 2; ++mt)
    for (int nt = 0; nt < 4; ++nt)
      for (int j = 0; j < 8; ++j) acc[mt][nt][j] = 0.0f;

  const __bf16* arow0 = A + (size_t)(mrow + lm) * K;
  const __bf16* arow1 = A + (size_t)(mrow + 16 + lm) * K;

  // A fragment: row (mrow + mt*16 + lm), K chunks {k0+8h..+7, k0+16+8h..+7}
  auto loadA = [&](int k0, v16bf a[2]) {
    {
      const __bf16* ap = arow0 + k0 + half * 8;
      union { v16bf v; v8bf h[2]; } u;
      u.h[0] = *(const v8bf*)ap;
      u.h[1] = *(const v8bf*)(ap + 16);
      a[0] = u.v;
    }
    {
      const __bf16* ap = arow1 + k0 + half * 8;
      union { v16bf v; v8bf h[2]; } u;
      u.h[0] = *(const v8bf*)ap;
      u.h[1] = *(const v8bf*)(ap + 16);
      a[1] = u.v;
    }
  };

  v16bf aCur[2], aNext[2];
  loadA(0, aCur);

  for (int k0 = 0; k0 < K; k0 += 32) {
    if (k0 + 32 < K) loadA(k0 + 32, aNext);   // pipeline next A slab
    for (int nt = 0; nt < 4; ++nt) {
      // B fragment from LDS: local row (nt*16+lm), K chunks {k0+8h..+7, k0+16+8h..+7}
      const __bf16* wp = &wtile[(nt * 16 + lm) * K + k0 + half * 8];
      union { v16bf v; v8bf h[2]; } u;
      u.h[0] = *(const v8bf*)wp;              // ds_load_b128
      u.h[1] = *(const v8bf*)(wp + 16);
      v16bf b = u.v;
      acc[0][nt] = __builtin_amdgcn_wmma_f32_16x16x32_bf16(
          false, aCur[0], false, b, (short)0, acc[0][nt], false, false);
      acc[1][nt] = __builtin_amdgcn_wmma_f32_16x16x32_bf16(
          false, aCur[1], false, b, (short)0, acc[1][nt], false, false);
    }
    aCur[0] = aNext[0];
    aCur[1] = aNext[1];
  }

  // Epilogue: C/D layout -> VGPR j holds (M = j + 8*half, N = lm) per tile.
  for (int nt = 0; nt < 4; ++nt) {
    int col = n0 + nt * 16 + lm;
    if (col >= N) continue;
    float bb = 0.0f;
    if (bias0) bb += bias0[col];
    if (bias1) bb += bias1[col];
    for (int mt = 0; mt < 2; ++mt)
      for (int j = 0; j < 8; ++j) {
        int row = mrow + mt * 16 + half * 8 + j;
        float v = acc[mt][nt][j] + bb;
        if (NT)
          __builtin_nontemporal_store(v, &Out[(size_t)row * N + col]);
        else
          Out[(size_t)row * N + col] = v;
      }
  }
}

// ---------------------------------------------------------------------------
// LSTM recurrence (one layer, one 1024-thread workgroup = 32 wave32s).
// gates_x already holds x@Wih^T + b_ih + b_hh.  Each wave owns 16 h-columns
// and computes all 4 gate tiles for both batch half-tiles (8 WMMA per k-step).
// h state double-buffered in LDS (bf16); c state lives in registers.
// ---------------------------------------------------------------------------
__global__ __launch_bounds__(1024, 1) void lstm_recurrence(
    const float*  __restrict__ gates_x, // [T*B, 4H]
    const __bf16* __restrict__ Whh,     // [4H, H] bf16
    const float*  __restrict__ h0,      // [B, H]
    const float*  __restrict__ c0,      // [B, H]
    __bf16* __restrict__ hs,            // [T*B, H] out (bf16, next layer A)
    float* __restrict__ hT,             // [B, H] out
    float* __restrict__ cT)             // [B, H] out
{
  __shared__ __align__(16) __bf16 hbuf[2][BB * HH];   // 2 * 32KB double buffer

  int tid  = threadIdx.x;
  int w    = tid >> 5;
  int lane = tid & 31;
  int half = lane >> 4;
  int lm   = lane & 15;
  int hcol = w * 16 + lm;                // this wave's h-column (0..511)

  for (int i = tid; i < BB * HH; i += 1024) hbuf[0][i] = f2bf(h0[i]);

  float cst[2][8];
  for (int mt = 0; mt < 2; ++mt)
    for (int j = 0; j < 8; ++j)
      cst[mt][j] = c0[(mt * 16 + half * 8 + j) * HH + hcol];

  __syncthreads();

  for (int t = 0; t < TT; ++t) {
    int cur = t & 1, nxt = cur ^ 1;

    // Init accumulators from the precomputed input projection (+biases).
    v8f acc[2][4];
    for (int mt = 0; mt < 2; ++mt)
      for (int g = 0; g < 4; ++g)
        for (int j = 0; j < 8; ++j)
          acc[mt][g][j] =
              gates_x[(size_t)(t * BB + mt * 16 + half * 8 + j) * (4 * HH) +
                      g * HH + hcol];

    // gates += h_{t-1} @ Whh^T  (A from LDS, B from L2-resident Whh)
    for (int k0 = 0; k0 < HH; k0 += 32) {
      v16bf a[2];
      for (int mt = 0; mt < 2; ++mt) {
        const __bf16* ap = &hbuf[cur][(mt * 16 + lm) * HH + k0 + half * 8];
        union { v16bf v; v8bf h[2]; } u;
        u.h[0] = *(const v8bf*)ap;           // ds_load_b128
        u.h[1] = *(const v8bf*)(ap + 16);
        a[mt] = u.v;
      }
      for (int g = 0; g < 4; ++g) {          // sequential B to limit VGPR live
        const __bf16* wp =
            Whh + (size_t)(g * HH + hcol) * HH + k0 + half * 8;
        union { v16bf v; v8bf h[2]; } u;
        u.h[0] = *(const v8bf*)wp;
        u.h[1] = *(const v8bf*)(wp + 16);
        v16bf b = u.v;
        acc[0][g] = __builtin_amdgcn_wmma_f32_16x16x32_bf16(
            false, a[0], false, b, (short)0, acc[0][g], false, false);
        acc[1][g] = __builtin_amdgcn_wmma_f32_16x16x32_bf16(
            false, a[1], false, b, (short)0, acc[1][g], false, false);
      }
    }

    // Gate activations (PyTorch order i,f,g,o), c/h update.
    for (int mt = 0; mt < 2; ++mt)
      for (int j = 0; j < 8; ++j) {
        float iv = sigmoidf_(acc[mt][0][j]);
        float fv = sigmoidf_(acc[mt][1][j]);
        float gv = tanhf(acc[mt][2][j]);
        float ov = sigmoidf_(acc[mt][3][j]);
        float cc = fv * cst[mt][j] + iv * gv;
        cst[mt][j] = cc;
        float hh = ov * tanhf(cc);
        int b = mt * 16 + half * 8 + j;
        __bf16 hb = f2bf(hh);
        hbuf[nxt][b * HH + hcol] = hb;
        hs[(size_t)(t * BB + b) * HH + hcol] = hb;
        if (t == TT - 1) {
          hT[b * HH + hcol] = hh;
          cT[b * HH + hcol] = cc;
        }
      }
    __syncthreads();   // all reads of hbuf[cur] done before next step writes it
  }
}

// ---------------------------------------------------------------------------
// Launcher
// ---------------------------------------------------------------------------
extern "C" void kernel_launch(void* const* d_in, const int* in_sizes, int n_in,
                              void* d_out, int out_size, void* d_ws,
                              size_t ws_size, hipStream_t stream) {
  (void)in_sizes; (void)n_in; (void)out_size; (void)ws_size;

  const int*   input = (const int*)  d_in[0];  // [T, B]
  const float* h0    = (const float*)d_in[1];  // [L, B, H]
  const float* c0    = (const float*)d_in[2];  // [L, B, H]
  const float* embW  = (const float*)d_in[3];  // [V, E]
  const float* Wih   = (const float*)d_in[4];  // [L, 4H, E]
  const float* Whh   = (const float*)d_in[5];  // [L, 4H, H]
  const float* bih   = (const float*)d_in[6];  // [L, 4H]
  const float* bhh   = (const float*)d_in[7];  // [L, 4H]
  const float* decW  = (const float*)d_in[8];  // [V, H]
  const float* decb  = (const float*)d_in[9];  // [V]
  float* out = (float*)d_out;

  const size_t TB = (size_t)TT * BB;

  // Workspace carve (~102 MB total)
  char* p = (char*)d_ws;
  auto carve = [&](size_t bytes) {
    char* r = p;
    p += (bytes + 255) & ~(size_t)255;
    return r;
  };
  __bf16* decW_bf = (__bf16*)carve((size_t)VV * HH * 2);        // ~51.5 MB
  __bf16* Wih_bf  = (__bf16*)carve((size_t)LL * 4 * HH * EE * 2);
  __bf16* Whh_bf  = (__bf16*)carve((size_t)LL * 4 * HH * HH * 2);
  __bf16* x_bf    = (__bf16*)carve(TB * EE * 2);
  __bf16* hs_bf   = (__bf16*)carve(TB * HH * 2);
  float*  gatesx  = (float*) carve(TB * 4 * HH * 4);            // 32 MB

  // 1) Precision casts (one pass; all weight reuse happens in bf16 from L2).
  {
    size_t n = (size_t)VV * HH;
    cast_f32_bf16<<<(unsigned)((n + 255) / 256), 256, 0, stream>>>(decW, decW_bf, n);
  }
  {
    size_t n = (size_t)LL * 4 * HH * EE;
    cast_f32_bf16<<<(unsigned)((n + 255) / 256), 256, 0, stream>>>(Wih, Wih_bf, n);
  }
  {
    size_t n = (size_t)LL * 4 * HH * HH;
    cast_f32_bf16<<<(unsigned)((n + 255) / 256), 256, 0, stream>>>(Whh, Whh_bf, n);
  }

  // 2) Embedding gather straight into bf16 A-matrix layout [T*B, E].
  {
    size_t n = TB * EE;
    embed_bf16<<<(unsigned)((n + 255) / 256), 256, 0, stream>>>(input, embW, x_bf);
  }

  const size_t logits_sz = TB * (size_t)VV;

  // 3) Per layer: batched input projection GEMM, then serial recurrence.
  for (int l = 0; l < LL; ++l) {
    const __bf16* Ain = (l == 0) ? x_bf : hs_bf;
    int M = (int)TB, N = 4 * HH, K = (l == 0) ? EE : HH;
    int blocks = (M / 32) * ((N + 63) / 64) / 8;   // MP*NP/8, exact
    gemm_bf16_wmma<false><<<blocks, 256, 0, stream>>>(
        Ain, Wih_bf + (size_t)l * 4 * HH * EE,
        bih + (size_t)l * 4 * HH, bhh + (size_t)l * 4 * HH,
        gatesx, M, N, K);

    lstm_recurrence<<<1, 1024, 0, stream>>>(
        gatesx, Whh_bf + (size_t)l * 4 * HH * HH,
        h0 + (size_t)l * BB * HH, c0 + (size_t)l * BB * HH,
        hs_bf,
        out + logits_sz + (size_t)l * BB * HH,
        out + logits_sz + (size_t)LL * BB * HH + (size_t)l * BB * HH);
  }

  // 4) Decoder GEMM (dominant): [4096,512] x [512,50257] + dec_b -> logits.
  //    Non-temporal stores: stream 824MB of logits to HBM, keep dec_W in L2.
  {
    int M = (int)TB, N = VV, K = HH;
    int blocks = (M / 32) * ((N + 63) / 64) / 8;
    gemm_bf16_wmma<true><<<blocks, 256, 0, stream>>>(
        hs_bf, decW_bf, decb, nullptr, out, M, N, K);
  }
}
